// HSTU_EncoderLayer_64476049048225
// MI455X (gfx1250) — matrix operand order repaired and checked by reading, not compiled, not run on hardware
//
#include <hip/hip_runtime.h>

// ---------------------------------------------------------------------------
// HSTU encoder layer for MI455X (gfx1250, wave32, WMMA 16x16x32 f16).
// ---------------------------------------------------------------------------

typedef __attribute__((ext_vector_type(16))) _Float16 v16h;
typedef __attribute__((ext_vector_type(8)))  _Float16 v8h;
typedef __attribute__((ext_vector_type(8)))  float    v8f;

#define DEV __device__ __forceinline__

static constexpr int S_   = 2048;
static constexpr int B_   = 2;
static constexpr int E_   = 1024;
static constexpr int H_   = 16;
static constexpr int D_   = 64;
static constexpr int R_   = S_ * B_;      // 4096 flattened (s,b) rows
static constexpr int N2E  = 2 * E_;       // 2048

DEV v8f wmma16(v16h a, v16h b, v8f c) {
  // D = A(16x32 f16) * B(32x16 f16) + C(16x16 f32)
  return __builtin_amdgcn_wmma_f32_16x16x32_f16(
      /*neg_a=*/false, a, /*neg_b=*/false, b,
      /*c_mod=*/(short)0, c, /*reuse_a=*/false, /*reuse_b=*/false);
}

// A-fragment (16x32, 16-bit): caller passes ptr = base + row(lane&15)*ld + k0 + (lane>>4)*8
// loads K = {off..off+7} and {off+16..off+23}; per ISA 16-bit A layout.
DEV v16h load_frag_a(const _Float16* p) {
  v8h lo = *(const v8h*)(p);
  v8h hi = *(const v8h*)(p + 16);
  v16h a;
#pragma unroll
  for (int i = 0; i < 8; ++i) { a[i] = lo[i]; a[i + 8] = hi[i]; }
  return a;
}

// B-fragment (32x16, 16-bit): caller passes ptr for column n = lane&15 with K-offset
// (lane>>4)*8 applied; loads K = {off..off+7} and {off+16..off+23} (matches A layout).
DEV v16h load_frag_b(const _Float16* p) {
  v8h lo = *(const v8h*)(p);
  v8h hi = *(const v8h*)(p + 16);
  v16h b;
#pragma unroll
  for (int i = 0; i < 8; ++i) { b[i] = lo[i]; b[i + 8] = hi[i]; }
  return b;
}

DEV float silu(float x) { return x / (1.0f + __expf(-x)); }

// ---------------------------------------------------------------------------
// Kernel 1: fp32 -> fp16 weight cast
// ---------------------------------------------------------------------------
__global__ void cast_f16_kernel(const float* __restrict__ in,
                                _Float16* __restrict__ out, int n) {
  int i = blockIdx.x * blockDim.x + threadIdx.x;
  if (i < n) out[i] = (_Float16)in[i];
}

// ---------------------------------------------------------------------------
// Kernel 2: LayerNorm over E=1024 per (s,b) row, emit fp16
// ---------------------------------------------------------------------------
__global__ __launch_bounds__(256) void ln_kernel(const float* __restrict__ src,
                                                 const float* __restrict__ g,
                                                 const float* __restrict__ bta,
                                                 _Float16* __restrict__ xn) {
  const int row = blockIdx.x;
  const int tid = threadIdx.x;
  const float* x = src + (size_t)row * E_;
  __shared__ float ssum[256];
  __shared__ float ssq[256];
  float v[4], s = 0.f, q = 0.f;
#pragma unroll
  for (int i = 0; i < 4; ++i) {
    v[i] = x[tid + i * 256];
    s += v[i];
    q += v[i] * v[i];
  }
  ssum[tid] = s; ssq[tid] = q;
  __syncthreads();
  for (int off = 128; off > 0; off >>= 1) {
    if (tid < off) { ssum[tid] += ssum[tid + off]; ssq[tid] += ssq[tid + off]; }
    __syncthreads();
  }
  const float mu  = ssum[0] * (1.0f / E_);
  const float var = ssq[0] * (1.0f / E_) - mu * mu;
  const float rs  = rsqrtf(var + 1e-5f);
#pragma unroll
  for (int i = 0; i < 4; ++i) {
    int c = tid + i * 256;
    xn[(size_t)row * E_ + c] = (_Float16)((v[i] - mu) * rs * g[c] + bta[c]);
  }
}

// ---------------------------------------------------------------------------
// Dense GEMM core: C[128x64 per block] = A[M x K] * BT[N x K]^T.
// Block = 128 thr (4 waves). Each wave owns a 32x64 C tile:
//   2 A-fragments + 4 shared B-fragments -> 8 WMMAs per K-step.
// B tile staged in LDS; next K-tile prefetched with global_prefetch.
// ---------------------------------------------------------------------------
struct GemmFrag {
  v8f acc[2][4];
};

DEV void gemm_core(const _Float16* __restrict__ A,
                   const _Float16* __restrict__ BT,
                   _Float16* sB, int m0, int n0, int lane, GemmFrag& fr) {
  constexpr int TS = 40;  // padded LDS row stride (halves)
  const int tid = threadIdx.x;
  const int m15 = lane & 15;
  const int hf  = lane >> 4;
  const _Float16* arow0 = A + (size_t)(m0 + m15) * E_ + hf * 8;
  const _Float16* arow1 = arow0 + (size_t)16 * E_;

  for (int k0 = 0; k0 < E_; k0 += 32) {
    // cooperative stage of BT tile (64 n-rows x 32 k) into LDS
    for (int c = tid; c < 256; c += 128) {
      int r = c >> 2, off = (c & 3) * 8;
      const _Float16* gp = BT + (size_t)(n0 + r) * E_ + k0 + off;
      *(v8h*)&sB[r * TS + off] = *(const v8h*)gp;
      if (k0 + 32 < E_) __builtin_prefetch(gp + 32, 0, 1);  // next K-tile
    }
    __syncthreads();
    v16h a0 = load_frag_a(arow0 + k0);
    v16h a1 = load_frag_a(arow1 + k0);
#pragma unroll
    for (int j = 0; j < 4; ++j) {
      v16h bf = load_frag_b(&sB[(j * 16 + m15) * TS + hf * 8]);
      fr.acc[0][j] = wmma16(a0, bf, fr.acc[0][j]);
      fr.acc[1][j] = wmma16(a1, bf, fr.acc[1][j]);
    }
    __syncthreads();
  }
}

// ---------------------------------------------------------------------------
// Kernel 3: proj = silu(xn @ Win^T + bin); split into qkv16 / u16.
// ---------------------------------------------------------------------------
__global__ __launch_bounds__(128) void gemm_in_kernel(
    const _Float16* __restrict__ A,   // xn16 (4096 x 1024)
    const _Float16* __restrict__ BT,  // wi16 (2048 x 1024), BT[n][k]
    const float* __restrict__ bias,   // 2048
    _Float16* __restrict__ qkv, _Float16* __restrict__ u) {
  __shared__ _Float16 sB[64 * 40];
  const int lane = threadIdx.x & 31;
  const int wave = threadIdx.x >> 5;
  const int m15  = lane & 15;
  const int hf   = lane >> 4;
  const int n0   = blockIdx.x * 64;
  const int m0   = blockIdx.y * 128 + wave * 32;

  GemmFrag fr = {};
  gemm_core(A, BT, sB, m0, n0, lane, fr);

#pragma unroll
  for (int mi = 0; mi < 2; ++mi) {
    const int mrow = m0 + mi * 16 + hf * 8;
#pragma unroll
    for (int j = 0; j < 4; ++j) {
      const int n = n0 + j * 16 + m15;
      const float bv = bias[n];
#pragma unroll
      for (int r = 0; r < 8; ++r) {
        float sv = silu(fr.acc[mi][j][r] + bv);
        size_t m = (size_t)(mrow + r);
        if (n < E_) qkv[m * E_ + n] = (_Float16)sv;
        else        u[m * E_ + (n - E_)] = (_Float16)sv;
      }
    }
  }
}

// ---------------------------------------------------------------------------
// Kernel 4: causal SiLU-attention + u-gating.
//   scores = Q K^T / 8 ; W = silu(scores) masked causally ; O = W V ; gated = O*u
// Q=K=V=qkv16. Block = 128 thr (4 waves), 64 query rows per block, one (b,h).
// ---------------------------------------------------------------------------
__global__ __launch_bounds__(128) void attn_kernel(
    const _Float16* __restrict__ qkv, const _Float16* __restrict__ u,
    _Float16* __restrict__ gated) {
  __shared__ _Float16 kvs[32 * 72];     // KV tile row-major  [t][d]
  __shared__ _Float16 kvts[64 * 40];    // KV tile transposed [d][t]
  __shared__ _Float16 sc[4][16 * 40];   // per-wave score tile (16 x 32)

  const int tid  = threadIdx.x;
  const int lane = tid & 31;
  const int wave = tid >> 5;
  const int m15  = lane & 15;
  const int hf   = lane >> 4;
  const int stripe = blockIdx.x;        // 0..31 (64 query rows each)
  const int bh = blockIdx.y;            // 0..31
  const int b  = bh & (B_ - 1);
  const int h  = bh >> 1;
  const int colbase = h * D_;
  const int s0 = stripe * 64 + wave * 16;

  // Q fragments (d = 0..31 and 32..63), loaded once
  const _Float16* qrow = qkv + ((size_t)(s0 + m15) * B_ + b) * E_ + colbase + hf * 8;
  v16h aq0 = load_frag_a(qrow);
  v16h aq1 = load_frag_a(qrow + 32);

  v8f o[4] = {};
  const int sMaxWave = s0 + 15;
  const int tEnd = stripe * 64 + 63;

  for (int t0 = 0; t0 <= tEnd; t0 += 32) {
    // stage KV tile (32 t-rows x 64 d) both row-major and transposed
    for (int c = tid; c < 256; c += 128) {
      int tl = c >> 3, doff = (c & 7) * 8;
      const _Float16* gp = qkv + ((size_t)(t0 + tl) * B_ + b) * E_ + colbase + doff;
      v8h v = *(const v8h*)gp;
      if (t0 + 32 <= tEnd) __builtin_prefetch(gp + (size_t)32 * B_ * E_, 0, 1);
      *(v8h*)&kvs[tl * 72 + doff] = v;
#pragma unroll
      for (int i = 0; i < 8; ++i) kvts[(doff + i) * 40 + tl] = v[i];
    }
    __syncthreads();

    if (t0 <= sMaxWave) {
      _Float16* scw = sc[wave];
#pragma unroll
      for (int nt = 0; nt < 2; ++nt) {
        v8f cs = {};
        const _Float16* bp0 = &kvs[(nt * 16 + m15) * 72 + hf * 8];
        cs = wmma16(aq0, load_frag_b(bp0), cs);        // d = 0..31
        cs = wmma16(aq1, load_frag_b(bp0 + 32), cs);   // d = 32..63
        const int tg = t0 + nt * 16 + m15;             // key index of this lane's column
        const int sgBase = s0 + hf * 8;
#pragma unroll
        for (int r = 0; r < 8; ++r) {
          float x  = cs[r] * 0.125f;                   // 1/sqrt(64)
          float wv = (tg <= sgBase + r) ? silu(x) : 0.0f;
          scw[(hf * 8 + r) * 40 + nt * 16 + m15] = (_Float16)wv;
        }
      }
      // reload score tile as an A-fragment (16 x 32 over t)
      v16h aw = load_frag_a(scw + m15 * 40 + hf * 8);
#pragma unroll
      for (int j = 0; j < 4; ++j) {
        const _Float16* bp = &kvts[(j * 16 + m15) * 40 + hf * 8];
        o[j] = wmma16(aw, load_frag_b(bp), o[j]);
      }
    }
    __syncthreads();
  }

  // gated epilogue: gated = O * u (u already silu'ed)
#pragma unroll
  for (int j = 0; j < 4; ++j) {
    const int d = colbase + j * 16 + m15;
#pragma unroll
    for (int r = 0; r < 8; ++r) {
      const int s = s0 + hf * 8 + r;
      size_t idx = ((size_t)s * B_ + b) * E_ + d;
      gated[idx] = (_Float16)(o[j][r] * (float)u[idx]);
    }
  }
}

// ---------------------------------------------------------------------------
// Kernel 5: out = src + gated @ Wout^T + bout.  C is 4096 x 1024, K = 1024.
// ---------------------------------------------------------------------------
__global__ __launch_bounds__(128) void gemm_out_kernel(
    const _Float16* __restrict__ A,   // gated16 (4096 x 1024)
    const _Float16* __restrict__ BT,  // wo16 (1024 x 1024)
    const float* __restrict__ bias,   // 1024
    const float* __restrict__ src,    // residual
    float* __restrict__ out) {
  __shared__ _Float16 sB[64 * 40];
  const int lane = threadIdx.x & 31;
  const int wave = threadIdx.x >> 5;
  const int m15  = lane & 15;
  const int hf   = lane >> 4;
  const int n0   = blockIdx.x * 64;
  const int m0   = blockIdx.y * 128 + wave * 32;

  GemmFrag fr = {};
  gemm_core(A, BT, sB, m0, n0, lane, fr);

#pragma unroll
  for (int mi = 0; mi < 2; ++mi) {
    const int mrow = m0 + mi * 16 + hf * 8;
#pragma unroll
    for (int j = 0; j < 4; ++j) {
      const int n = n0 + j * 16 + m15;
      const float bv = bias[n];
#pragma unroll
      for (int r = 0; r < 8; ++r) {
        size_t idx = (size_t)(mrow + r) * E_ + n;
        out[idx] = src[idx] + fr.acc[mi][j][r] + bv;
      }
    }
  }
}

// ---------------------------------------------------------------------------
// Host-side launcher
// ---------------------------------------------------------------------------
extern "C" void kernel_launch(void* const* d_in, const int* in_sizes, int n_in,
                              void* d_out, int out_size, void* d_ws, size_t ws_size,
                              hipStream_t stream) {
  const float* src = (const float*)d_in[0];
  // d_in[1] = src_mask (causal tril) -> enforced analytically in attn_kernel
  const float* wi  = (const float*)d_in[2];
  const float* bi  = (const float*)d_in[3];
  const float* wo  = (const float*)d_in[4];
  const float* bo  = (const float*)d_in[5];
  const float* lng = (const float*)d_in[6];
  const float* lnb = (const float*)d_in[7];
  float* out = (float*)d_out;

  // workspace layout (bytes)
  char* ws = (char*)d_ws;
  _Float16* xn16    = (_Float16*)(ws);                     // 4096x1024 f16 =  8 MB
  _Float16* qkv16   = (_Float16*)(ws + (8u  << 20));       // 4096x1024 f16 =  8 MB
  _Float16* u16     = (_Float16*)(ws + (16u << 20));       // 4096x1024 f16 =  8 MB
  _Float16* gated16 = (_Float16*)(ws + (24u << 20));       // 4096x1024 f16 =  8 MB
  _Float16* wi16    = (_Float16*)(ws + (32u << 20));       // 2048x1024 f16 =  4 MB
  _Float16* wo16    = (_Float16*)(ws + (36u << 20));       // 1024x1024 f16 =  2 MB

  // 1) weight casts
  {
    int n1 = N2E * E_;   // 2,097,152
    int n2 = E_ * E_;    // 1,048,576
    cast_f16_kernel<<<(n1 + 255) / 256, 256, 0, stream>>>(wi, wi16, n1);
    cast_f16_kernel<<<(n2 + 255) / 256, 256, 0, stream>>>(wo, wo16, n2);
  }
  // 2) layernorm
  ln_kernel<<<R_, 256, 0, stream>>>(src, lng, lnb, xn16);
  // 3) in-projection + silu + split
  gemm_in_kernel<<<dim3(N2E / 64, R_ / 128), 128, 0, stream>>>(xn16, wi16, bi, qkv16, u16);
  // 4) causal silu attention + gating
  attn_kernel<<<dim3(S_ / 64, B_ * H_), 128, 0, stream>>>(qkv16, u16, gated16);
  // 5) out-projection + bias + residual
  gemm_out_kernel<<<dim3(E_ / 64, R_ / 128), 128, 0, stream>>>(gated16, wo16, bo, src, out);
}
